// Block_73890617361064
// MI455X (gfx1250) — matrix operand order repaired and checked by reading, not compiled, not run on hardware
//
#include <hip/hip_runtime.h>
#include <hip/hip_bf16.h>
#include <math.h>

// ---------------------------------------------------------------------------
// Types for CDNA5 WMMA (wave32): V_WMMA_F32_16X16X32_BF16
// ---------------------------------------------------------------------------
typedef __attribute__((ext_vector_type(16))) __bf16 v16bf;
typedef __attribute__((ext_vector_type(8)))  float  v8f;
typedef __attribute__((ext_vector_type(4)))  unsigned int u32x4;

union Frag16 {             // 32 bytes = 16 bf16 = 8 VGPRs
    v16bf v;
    u32x4 u[2];
};

__device__ __forceinline__ v8f wmma_bf16(v16bf a, v16bf b, v8f c) {
    // (neg_a, A, neg_b, B, c_mod, C, reuse_a, reuse_b)
    return __builtin_amdgcn_wmma_f32_16x16x32_bf16(false, a, false, b,
                                                   (short)0, c, false, false);
}

__device__ __forceinline__ v8f v8f_zero() {
    v8f z;
#pragma unroll
    for (int i = 0; i < 8; ++i) z[i] = 0.0f;
    return z;
}

// ---- Fused DPP16 butterfly reductions (1 VALU op per step) ----------------
// DPP is a src0 modifier on VOP2: v_max_num_f32_dpp d, dpp(x), x
__device__ __forceinline__ float row16_max(float x) {
    float t;
    asm("v_max_num_f32_dpp %0, %1, %1 quad_perm:[1,0,3,2] row_mask:0xf bank_mask:0xf bound_ctrl:1"
        : "=v"(t) : "v"(x));
    asm("v_max_num_f32_dpp %0, %1, %1 quad_perm:[2,3,0,1] row_mask:0xf bank_mask:0xf bound_ctrl:1"
        : "=v"(x) : "v"(t));
    asm("v_max_num_f32_dpp %0, %1, %1 row_half_mirror row_mask:0xf bank_mask:0xf bound_ctrl:1"
        : "=v"(t) : "v"(x));
    asm("v_max_num_f32_dpp %0, %1, %1 row_mirror row_mask:0xf bank_mask:0xf bound_ctrl:1"
        : "=v"(x) : "v"(t));
    return x;
}
__device__ __forceinline__ float row16_sum(float x) {
    float t;
    asm("v_add_f32_dpp %0, %1, %1 quad_perm:[1,0,3,2] row_mask:0xf bank_mask:0xf bound_ctrl:1"
        : "=v"(t) : "v"(x));
    asm("v_add_f32_dpp %0, %1, %1 quad_perm:[2,3,0,1] row_mask:0xf bank_mask:0xf bound_ctrl:1"
        : "=v"(x) : "v"(t));
    asm("v_add_f32_dpp %0, %1, %1 row_half_mirror row_mask:0xf bank_mask:0xf bound_ctrl:1"
        : "=v"(t) : "v"(x));
    asm("v_add_f32_dpp %0, %1, %1 row_mirror row_mask:0xf bank_mask:0xf bound_ctrl:1"
        : "=v"(x) : "v"(t));
    return x;
}

// ---- CDNA5 async global -> LDS (ASYNCcnt path), 16B per lane --------------
__device__ __forceinline__ void async_b128(const void* gptr, void* lptr) {
    unsigned lds = (unsigned)(unsigned long long)lptr;           // low 32 = LDS offset
    unsigned long long ga = (unsigned long long)gptr;
    asm volatile("global_load_async_to_lds_b128 %0, %1, off"
                 :: "v"(lds), "v"(ga) : "memory");
}
__device__ __forceinline__ void wait_async0() {
    asm volatile("s_wait_asynccnt 0x0" ::: "memory");
}

// ---------------------------------------------------------------------------
// Elementwise helpers
// ---------------------------------------------------------------------------
__global__ __launch_bounds__(256) void cast_f32_bf16(const float* __restrict__ in,
                                                     __hip_bfloat16* __restrict__ out,
                                                     size_t n) {
    size_t i = (size_t)blockIdx.x * blockDim.x + threadIdx.x;
    if (i < n) out[i] = __float2bfloat16(in[i]);
}

// w[o,i] = (1/2) * sum_t (coef[0,t]+coef[1,t]) * tpl[t,o,i]   -> bf16
__global__ __launch_bounds__(256) void bank_combine(const float* __restrict__ tpl,
                                                    const float* __restrict__ coef,
                                                    __hip_bfloat16* __restrict__ w,
                                                    size_t sz, int T) {
    __shared__ float s[32];
    if ((int)threadIdx.x < T)
        s[threadIdx.x] = 0.5f * (coef[threadIdx.x] + coef[T + threadIdx.x]);
    __syncthreads();
    size_t i = (size_t)blockIdx.x * blockDim.x + threadIdx.x;
    if (i < sz) {
        float acc = 0.0f;
        for (int t = 0; t < T; ++t) acc += s[t] * tpl[(size_t)t * sz + i];
        w[i] = __float2bfloat16(acc);
    }
}

// Row LayerNorm over C, fp32 in -> bf16 out. One block (256 thr / 8 waves) per row.
__global__ __launch_bounds__(256) void layernorm_bf16(const float* __restrict__ x,
                                                      const float* __restrict__ g,
                                                      const float* __restrict__ b,
                                                      __hip_bfloat16* __restrict__ out,
                                                      int C) {
    const size_t row = blockIdx.x;
    const float* xr = x + row * (size_t)C;
    float s = 0.0f, sq = 0.0f;
    for (int i = threadIdx.x; i < C; i += blockDim.x) {
        float v = xr[i];
        s += v;
        sq += v * v;
    }
    // 16-lane DPP reduce, then one cross-half step via shfl_xor(16)
    s = row16_sum(s);  sq = row16_sum(sq);
    s += __shfl_xor(s, 16, 32);
    sq += __shfl_xor(sq, 16, 32);
    __shared__ float rs[8], rq[8];
    int wave = threadIdx.x >> 5, lane = threadIdx.x & 31;
    if (lane == 0) { rs[wave] = s; rq[wave] = sq; }
    __syncthreads();
    float ts = 0.0f, tq = 0.0f;
#pragma unroll
    for (int i = 0; i < 8; ++i) { ts += rs[i]; tq += rq[i]; }
    float mean = ts / (float)C;
    float var  = tq / (float)C - mean * mean;
    float inv  = rsqrtf(var + 1e-5f);
    for (int i = threadIdx.x; i < C; i += blockDim.x)
        out[row * (size_t)C + i] = __float2bfloat16((xr[i] - mean) * inv * g[i] + b[i]);
}

// ---------------------------------------------------------------------------
// LDS-tiled WMMA bf16 GEMM:  out[M,N] = A[M,K] * W[N,K]^T  (+bias)(gelu)(+res)
// Block: 256 thr (8 waves). Tile 128(M) x 128(N) x 32(K).
// Double-buffered LDS; tiles streamed with global_load_async_to_lds_b128 so
// the async DMA for tile k+1 overlaps the 8 WMMAs/wave of tile k.
// ---------------------------------------------------------------------------
#define BM 128
#define BN 128
#define BK 32
#define LDK (BK + 8)   // pad: keep 16B row alignment, dodge bank conflicts

template <bool OUT_BF16, bool HAS_BIAS, bool HAS_RES, bool DO_GELU>
__global__ __launch_bounds__(256) void gemm_wmma(const __hip_bfloat16* __restrict__ A,
                                                 const __hip_bfloat16* __restrict__ W,
                                                 const float* __restrict__ bias,
                                                 const float* __restrict__ res,
                                                 void* __restrict__ outp,
                                                 int M, int N, int K) {
    __shared__ __hip_bfloat16 As[2][BM][LDK];
    __shared__ __hip_bfloat16 Ws[2][BN][LDK];

    const int tid = threadIdx.x;
    const int wave = tid >> 5, lane = tid & 31;
    const int half = lane >> 4, l16 = lane & 15;
    const int m0 = blockIdx.y * BM;
    const int n0 = blockIdx.x * BN;

    v8f acc[8];
#pragma unroll
    for (int t = 0; t < 8; ++t) acc[t] = v8f_zero();

    const int r = tid >> 1;            // 0..127: staged row
    const int c = (tid & 1) * 16;      // 0 or 16 (each thread stages 32B/row/matrix)

    auto stage = [&](int buf, int k0) {
        const __hip_bfloat16* ap = A + (size_t)(m0 + r) * K + k0 + c;
        const __hip_bfloat16* wp = W + (size_t)(n0 + r) * K + k0 + c;
        async_b128(ap,     &As[buf][r][c]);
        async_b128(ap + 8, &As[buf][r][c + 8]);
        async_b128(wp,     &Ws[buf][r][c]);
        async_b128(wp + 8, &Ws[buf][r][c + 8]);
    };

    const int nk = K / BK;
    stage(0, 0);
    wait_async0();
    __syncthreads();

    for (int kb = 0; kb < nk; ++kb) {
        const int cur = kb & 1;
        if (kb + 1 < nk) stage(cur ^ 1, (kb + 1) * BK);   // overlap DMA w/ WMMA

        // ---- A fragment for this wave's 16 rows (ISA 16-bit A 16x32 layout) ----
        Frag16 af;
        {
            const __hip_bfloat16* pr = &As[cur][wave * 16 + l16][0];
            af.u[0] = *(const u32x4*)&pr[half ? 8 : 0];
            af.u[1] = *(const u32x4*)&pr[16 + (half ? 8 : 0)];
        }
#pragma unroll
        for (int t = 0; t < 8; ++t) {
            // B fragment: lanes 0-15 hold K=0..7,16..23 of column l16, lanes 16-31 K=8..15,24..31
            Frag16 bf;
            const __hip_bfloat16* pr = &Ws[cur][t * 16 + l16][0];
            bf.u[0] = *(const u32x4*)&pr[half ? 8 : 0];
            bf.u[1] = *(const u32x4*)&pr[16 + (half ? 8 : 0)];
            acc[t] = wmma_bf16(af.v, bf.v, acc[t]);
        }
        wait_async0();          // next tile landed (own wave's writes)
        __syncthreads();        // everyone's writes visible; frags already consumed
    }

    // ---- epilogue: D layout = VGPR v -> row v + 8*half, lane l16 -> col ----
#pragma unroll
    for (int t = 0; t < 8; ++t) {
        const int col = n0 + t * 16 + l16;
        float bv = HAS_BIAS ? bias[col] : 0.0f;
#pragma unroll
        for (int v = 0; v < 8; ++v) {
            const int row = m0 + wave * 16 + v + 8 * half;
            float val = acc[t][v] + bv;
            if (DO_GELU) val = 0.5f * val * (1.0f + erff(val * 0.70710678118654752f));
            if (HAS_RES) val += res[(size_t)row * N + col];
            if (OUT_BF16)
                ((__hip_bfloat16*)outp)[(size_t)row * N + col] = __float2bfloat16(val);
            else
                ((float*)outp)[(size_t)row * N + col] = val;
        }
    }
}

// ---------------------------------------------------------------------------
// Flash attention, all matmuls on WMMA.  qkv: [B,N,3C] bf16, out: [B,N,C] bf16
// grid = (N/128, H, B), block = 256 (8 waves); wave owns 16 q-rows, D = 64.
// K tile streamed via async-to-LDS; softmax row-reduce on fused DPP16.
// ---------------------------------------------------------------------------
__global__ __launch_bounds__(256) void flash_attn(const __hip_bfloat16* __restrict__ qkv,
                                                  __hip_bfloat16* __restrict__ o,
                                                  int B, int N, int C, int H) {
    const int D = 64;
    const int b = blockIdx.z, h = blockIdx.y, qb = blockIdx.x * 128;
    const int tid = threadIdx.x;
    const int wave = tid >> 5, lane = tid & 31;
    const int half = lane >> 4, l16 = lane & 15;
    const size_t C3 = (size_t)3 * C;

    __shared__ __hip_bfloat16 Ks[32][64];        // K tile, row-major over d
    __shared__ __hip_bfloat16 Vt[64][40];        // V tile transposed [d][key], padded
    __shared__ __hip_bfloat16 Ps[8][16][32];     // per-wave P tile (m x 32 keys)

    // ---- Q fragments (held in registers for whole kernel): 16 rows x 64 d ----
    Frag16 aq[2];
    {
        const int row = qb + wave * 16 + l16;
        const __hip_bfloat16* p = qkv + ((size_t)b * N + row) * C3 + h * D;
#pragma unroll
        for (int kc = 0; kc < 2; ++kc) {
            aq[kc].u[0] = *(const u32x4*)&p[kc * 32 + (half ? 8 : 0)];
            aq[kc].u[1] = *(const u32x4*)&p[kc * 32 + 16 + (half ? 8 : 0)];
        }
    }

    float m_run[8], l_run[8];
    v8f oacc[4];
#pragma unroll
    for (int v = 0; v < 8; ++v) { m_run[v] = -INFINITY; l_run[v] = 0.0f; }
#pragma unroll
    for (int fc = 0; fc < 4; ++fc) oacc[fc] = v8f_zero();

    const float scale = 0.125f;   // 1/sqrt(64)
    const int kr = tid >> 3;            // 0..31  staged key row
    const int kd = (tid & 7) * 8;       // 0..56  staged d offset

    for (int j0 = 0; j0 < N; j0 += 32) {
        // ---- stage K async -> LDS; V loaded + transposed via DS scatter ----
        const __hip_bfloat16* kp = qkv + ((size_t)b * N + j0 + kr) * C3 + C + h * D + kd;
        async_b128(kp, &Ks[kr][kd]);
        const __hip_bfloat16* vp = qkv + ((size_t)b * N + j0 + kr) * C3 + 2 * C + h * D + kd;
        u32x4 vv = *(const u32x4*)vp;
        const __hip_bfloat16* ve = (const __hip_bfloat16*)&vv;
#pragma unroll
        for (int i = 0; i < 8; ++i) Vt[kd + i][kr] = ve[i];
        wait_async0();
        __syncthreads();

        // ---- S = Q*K^T for two 16-key subtiles (2 WMMAs each over D=64) ----
        v8f s[2];
#pragma unroll
        for (int kt = 0; kt < 2; ++kt) {
            v8f cacc = v8f_zero();
#pragma unroll
            for (int kc = 0; kc < 2; ++kc) {
                Frag16 bk;
                const __hip_bfloat16* pr = &Ks[kt * 16 + l16][kc * 32];
                bk.u[0] = *(const u32x4*)&pr[half ? 8 : 0];
                bk.u[1] = *(const u32x4*)&pr[16 + (half ? 8 : 0)];
                cacc = wmma_bf16(aq[kc].v, bk.v, cacc);
            }
            s[kt] = cacc;
        }

        // ---- online softmax; row v+8*half lives in one VGPR over a 16-lane half.
        //      Row reduction = fused DPP16 butterfly (1 VALU op per step).
#pragma unroll
        for (int v = 0; v < 8; ++v) {
            float a0 = s[0][v] * scale, a1 = s[1][v] * scale;
            float t = row16_max(fmaxf(a0, a1));
            float mn = fmaxf(m_run[v], t);
            float f  = __expf(m_run[v] - mn);
            float p0 = __expf(a0 - mn), p1 = __expf(a1 - mn);
            float rs = row16_sum(p0 + p1);
            l_run[v] = l_run[v] * f + rs;
            m_run[v] = mn;
#pragma unroll
            for (int fc = 0; fc < 4; ++fc) oacc[fc][v] *= f;
            const int mrow = v + 8 * half;
            Ps[wave][mrow][l16]      = __float2bfloat16(p0);
            Ps[wave][mrow][16 + l16] = __float2bfloat16(p1);
        }

        // ---- restage P into A-fragment layout; O += P * V (4 WMMAs) ----
        Frag16 pa;
        {
            const __hip_bfloat16* pr = &Ps[wave][l16][0];
            pa.u[0] = *(const u32x4*)&pr[half ? 8 : 0];
            pa.u[1] = *(const u32x4*)&pr[16 + (half ? 8 : 0)];
        }
#pragma unroll
        for (int fc = 0; fc < 4; ++fc) {
            Frag16 bv;
            const __hip_bfloat16* vr = &Vt[fc * 16 + l16][0];
            bv.u[0] = *(const u32x4*)&vr[half ? 8 : 0];
            bv.u[1] = *(const u32x4*)&vr[16 + (half ? 8 : 0)];
            oacc[fc] = wmma_bf16(pa.v, bv.v, oacc[fc]);
        }
        __syncthreads();
    }

    // ---- normalize and write out ----
#pragma unroll
    for (int fc = 0; fc < 4; ++fc)
#pragma unroll
        for (int v = 0; v < 8; ++v) {
            const int mrow = v + 8 * half;
            const int row = qb + wave * 16 + mrow;
            float val = oacc[fc][v] / l_run[v];
            o[((size_t)b * N + row) * C + h * D + fc * 16 + l16] = __float2bfloat16(val);
        }
}

// ---------------------------------------------------------------------------
// Host-side orchestration
// ---------------------------------------------------------------------------
extern "C" void kernel_launch(void* const* d_in, const int* in_sizes, int n_in,
                              void* d_out, int out_size, void* d_ws, size_t ws_size,
                              hipStream_t stream) {
    (void)in_sizes; (void)n_in; (void)out_size; (void)ws_size;
    const int B = 8, N = 1024, C = 1024, H = 16, T = 16, Hid = 4096;
    const int M = B * N;                  // 8192 token rows

    const float* x        = (const float*)d_in[0];
    const float* ln1_g    = (const float*)d_in[1];
    const float* ln1_b    = (const float*)d_in[2];
    const float* qkv_w    = (const float*)d_in[3];
    const float* proj_w   = (const float*)d_in[4];
    const float* proj_b   = (const float*)d_in[5];
    const float* ln2_g    = (const float*)d_in[6];
    const float* ln2_b    = (const float*)d_in[7];
    const float* tpl1     = (const float*)d_in[8];
    const float* coef1    = (const float*)d_in[9];
    const float* bias1    = (const float*)d_in[10];
    const float* tpl2     = (const float*)d_in[11];
    const float* coef2    = (const float*)d_in[12];
    const float* bias2    = (const float*)d_in[13];
    float* out            = (float*)d_out;

    // workspace layout (bytes), 256B aligned; total ~216 MB
    char* ws = (char*)d_ws;
    size_t off = 0;
    auto take = [&](size_t bytes) { char* p = ws + off; off += (bytes + 255) & ~(size_t)255; return p; };
    __hip_bfloat16* h_bf    = (__hip_bfloat16*)take((size_t)M * C * 2);        // LN1 out
    __hip_bfloat16* qkvw_bf = (__hip_bfloat16*)take((size_t)3 * C * C * 2);
    __hip_bfloat16* projw_bf= (__hip_bfloat16*)take((size_t)C * C * 2);
    __hip_bfloat16* w1_bf   = (__hip_bfloat16*)take((size_t)Hid * C * 2);
    __hip_bfloat16* w2_bf   = (__hip_bfloat16*)take((size_t)C * Hid * 2);
    __hip_bfloat16* qkv_bf  = (__hip_bfloat16*)take((size_t)M * 3 * C * 2);
    __hip_bfloat16* o_bf    = (__hip_bfloat16*)take((size_t)M * C * 2);
    float*          x1      = (float*)take((size_t)M * C * 4);                 // x + attn
    __hip_bfloat16* h2_bf   = (__hip_bfloat16*)take((size_t)M * C * 2);        // LN2 out
    __hip_bfloat16* y_bf    = (__hip_bfloat16*)take((size_t)M * Hid * 2);      // gelu(mlp1)

    // 1) weight casts + dynamic weight banks (memory-bound, ~560 MB total)
    {
        size_t n = (size_t)3 * C * C;
        cast_f32_bf16<<<dim3((n + 255) / 256), dim3(256), 0, stream>>>(qkv_w, qkvw_bf, n);
        n = (size_t)C * C;
        cast_f32_bf16<<<dim3((n + 255) / 256), dim3(256), 0, stream>>>(proj_w, projw_bf, n);
        n = (size_t)Hid * C;
        bank_combine<<<dim3((n + 255) / 256), dim3(256), 0, stream>>>(tpl1, coef1, w1_bf, n, T);
        bank_combine<<<dim3((n + 255) / 256), dim3(256), 0, stream>>>(tpl2, coef2, w2_bf, n, T);
    }

    // 2) LN1 -> bf16
    layernorm_bf16<<<dim3(M), dim3(256), 0, stream>>>(x, ln1_g, ln1_b, h_bf, C);

    // 3) QKV GEMM: [8192,1024] x [3072,1024]^T -> bf16 [8192,3072]
    gemm_wmma<true, false, false, false>
        <<<dim3(3 * C / BN, M / BM), dim3(256), 0, stream>>>(
            h_bf, qkvw_bf, nullptr, nullptr, qkv_bf, M, 3 * C, C);

    // 4) flash attention -> o_bf [8192,1024]
    flash_attn<<<dim3(N / 128, H, B), dim3(256), 0, stream>>>(qkv_bf, o_bf, B, N, C, H);

    // 5) proj GEMM + bias + residual(x) -> x1 fp32
    gemm_wmma<false, true, true, false>
        <<<dim3(C / BN, M / BM), dim3(256), 0, stream>>>(
            o_bf, projw_bf, proj_b, x, x1, M, C, C);

    // 6) LN2 -> bf16
    layernorm_bf16<<<dim3(M), dim3(256), 0, stream>>>(x1, ln2_g, ln2_b, h2_bf, C);

    // 7) MLP1 + bias1 + exact GELU -> y_bf [8192,4096]
    gemm_wmma<true, true, false, true>
        <<<dim3(Hid / BN, M / BM), dim3(256), 0, stream>>>(
            h2_bf, w1_bf, bias1, nullptr, y_bf, M, Hid, C);

    // 8) MLP2 + bias2 + residual(x1) -> d_out fp32
    gemm_wmma<false, true, true, false>
        <<<dim3(C / BN, M / BM), dim3(256), 0, stream>>>(
            y_bf, w2_bf, bias2, x1, out, M, C, Hid);
}
